// LSTM_ATTN_Decoder_69982197121768
// MI455X (gfx1250) — matrix operand order repaired and checked by reading, not compiled, not run on hardware
//
#include <hip/hip_runtime.h>

// Problem constants (from reference)
#define Bd 16
#define Sd 128
#define Td 64
#define Ed 768
#define Hd 1024
#define Vd 32000
#define Gd 4096   // 4*H

typedef unsigned short u16;
typedef __attribute__((ext_vector_type(16))) __bf16 v16bf;
typedef __attribute__((ext_vector_type(8)))  float  v8f;
typedef unsigned int u32x4 __attribute__((ext_vector_type(4)));
typedef unsigned int v4u __attribute__((ext_vector_type(4)));
typedef int v8i __attribute__((ext_vector_type(8)));
typedef int v4i __attribute__((ext_vector_type(4)));

#if defined(__has_builtin)
# if __has_builtin(__builtin_amdgcn_tensor_load_to_lds) && __has_builtin(__builtin_amdgcn_s_wait_tensorcnt)
#  define USE_TDM 1
# endif
#endif
#ifndef USE_TDM
# define USE_TDM 0
#endif

union Pack16 { v16bf v; u32x4 q[2]; };

__device__ __forceinline__ u16 f2bf(float f) {
  unsigned int u = __float_as_uint(f);
  u += 0x7FFFu + ((u >> 16) & 1u);   // round-to-nearest-even
  return (u16)(u >> 16);
}
__device__ __forceinline__ float sigm(float x) { return 1.f / (1.f + __expf(-x)); }

// ---------------------------------------------------------------------------
// Wave-level 16x16 tile accumulate: D += A(16xK) * B tile, bf16 -> f32.
//   A frag: lane&15 = row m, halves K {ka..ka+7, ka+16..ka+23}, ka=(lane>>4)*8
//   B frag: lane&15 = col n, halves K {kb..kb+7, kb+16..kb+23}, kb=(lane>>4)*8
// Bt is the (N,K)-row-major weight tile base (row n == output column n).
// ---------------------------------------------------------------------------
__device__ __forceinline__ v8f wmma_tile_acc(const u16* A, int lda,
                                             const u16* Bt, int ldb,
                                             int K, v8f acc, int lane)
{
  const int m  = lane & 15;
  const int ka = (lane >> 4) << 3;
  const int kb = (lane >> 4) << 3;
  const u16* ap = A  + (size_t)m * lda + ka;
  const u16* bp = Bt + (size_t)m * ldb + kb;
  for (int kk = 0; kk < K; kk += 32) {
    Pack16 a, b;
    a.q[0] = *(const u32x4*)(ap + kk);
    a.q[1] = *(const u32x4*)(ap + kk + 16);
    b.q[0] = *(const u32x4*)(bp + kk);
    b.q[1] = *(const u32x4*)(bp + kk + 16);
    acc = __builtin_amdgcn_wmma_f32_16x16x32_bf16(false, a.v, false, b.v,
                                                  (short)0, acc, false, false);
  }
  return acc;
}

// 4-gate variant: one A-fragment load feeds 4 WMMAs (i,f,g,o weight tiles at
// gstride apart). Cuts A traffic 4x vs 4 independent passes.
__device__ __forceinline__ void wmma4_acc(const u16* A, int lda,
                                          const u16* Bt0, int ldb, size_t gstride,
                                          int K, v8f acc[4], int lane)
{
  const int m  = lane & 15;
  const int ka = (lane >> 4) << 3;
  const int kb = (lane >> 4) << 3;
  const u16* ap = A   + (size_t)m * lda + ka;
  const u16* bp = Bt0 + (size_t)m * ldb + kb;
  for (int kk = 0; kk < K; kk += 32) {
    Pack16 a;
    a.q[0] = *(const u32x4*)(ap + kk);
    a.q[1] = *(const u32x4*)(ap + kk + 16);
#pragma unroll
    for (int g = 0; g < 4; ++g) {
      Pack16 b;
      b.q[0] = *(const u32x4*)(bp + g * gstride + kk);
      b.q[1] = *(const u32x4*)(bp + g * gstride + kk + 16);
      acc[g] = __builtin_amdgcn_wmma_f32_16x16x32_bf16(false, a.v, false, b.v,
                                                       (short)0, acc[g], false, false);
    }
  }
}

// D layout: VGPR r -> M = r + 8*(lane>=16), N = lane&15
__device__ __forceinline__ void store_tile(v8f acc, float* C, int ldc,
                                           int m0, int n0, const float* bias, int lane)
{
  const int col = n0 + (lane & 15);
  const int rb  = (lane >> 4) << 3;
  const float bv = bias ? bias[col] : 0.f;
#pragma unroll
  for (int r = 0; r < 8; ++r)
    C[(size_t)(m0 + rb + r) * ldc + col] = acc[r] + bv;
}

// ---------------------------------------------------------------------------
// Generic GEMM: C(MxN) = Abf(MxK) @ Wbf(N,K)^T + bias. 256 thr = 8 waves,
// grid = (M/16, N/128). Used for x_proj precompute and h2 @ attn_Wh.
// ---------------------------------------------------------------------------
__global__ void k_gemm_bias(const u16* __restrict__ A, int lda,
                            const u16* __restrict__ Bw, int ldb,
                            const float* __restrict__ bias,
                            float* __restrict__ C, int ldc, int K)
{
  const int lane = threadIdx.x & 31, wave = threadIdx.x >> 5;
  const int m0 = blockIdx.x * 16;
  const int n0 = (blockIdx.y * 8 + wave) * 16;
  v8f acc = {};
  acc = wmma_tile_acc(A + (size_t)m0 * lda, lda, Bw + (size_t)n0 * ldb, ldb, K, acc, lane);
  store_tile(acc, C, ldc, m0, n0, bias, lane);
}

// ---------------------------------------------------------------------------
// Fused LSTM layer: for its 16 H-columns each wave computes all four gate
// tiles (WMMA) then applies the pointwise cell update in-register.
//   gates = [A1a|A1b](16xK1) @ Wih^T + A2(16x1024) @ Whh^T + bih + bhh
//   c' = sig(f)*c + sig(i)*tanh(g);  h' = sig(o)*tanh(c')  -> c, hbf(bf16)
// hin (h(t-1)) and hout (h(t)) are distinct buffers (ping-pong) to avoid the
// cross-block RAW race; c is element-in-place (each lane owns its element).
// grid = Hd/128 blocks x 256 thr (8 waves -> 64 column tiles).
// ---------------------------------------------------------------------------
__global__ void k_lstm_fused(const u16* __restrict__ A1a, int K1a,
                             const u16* __restrict__ A1b, int K1b,
                             const u16* __restrict__ B1, int ldb1,
                             const u16* __restrict__ hin,
                             const u16* __restrict__ B2,
                             const float* __restrict__ bih,
                             const float* __restrict__ bhh,
                             float* __restrict__ c,
                             u16* __restrict__ hout)
{
  const int lane = threadIdx.x & 31, wave = threadIdx.x >> 5;
  const int n0 = (blockIdx.x * 8 + wave) * 16;       // column tile within H
  v8f acc[4] = {{}, {}, {}, {}};
  const size_t gs1 = (size_t)Hd * ldb1;              // gate stride in B1
  const size_t gs2 = (size_t)Hd * Hd;                // gate stride in B2
  const u16* B1t = B1 + (size_t)n0 * ldb1;
  wmma4_acc(A1a, K1a, B1t, ldb1, gs1, K1a, acc, lane);
  if (A1b) wmma4_acc(A1b, K1b, B1t + K1a, ldb1, gs1, K1b, acc, lane);
  wmma4_acc(hin, Hd, B2 + (size_t)n0 * Hd, Hd, gs2, Hd, acc, lane);

  const int col = n0 + (lane & 15);
  const int rb  = (lane >> 4) << 3;
  float bv[4];
#pragma unroll
  for (int g = 0; g < 4; ++g) bv[g] = bih[g * Hd + col] + bhh[g * Hd + col];
#pragma unroll
  for (int r = 0; r < 8; ++r) {
    const size_t si = (size_t)(rb + r) * Hd + col;   // (batch row, H col)
    const float ig = acc[0][r] + bv[0];
    const float fg = acc[1][r] + bv[1];
    const float gg = acc[2][r] + bv[2];
    const float og = acc[3][r] + bv[3];
    const float cn = sigm(fg) * c[si] + sigm(ig) * tanhf(gg);
    const float hn = sigm(og) * tanhf(cn);
    c[si] = cn;
    hout[si] = f2bf(hn);
  }
}

// ---------------------------------------------------------------------------
// Vocab projection: logits(16x32000) = h2bf @ fcWbf(32000,1024)^T + fc_b
// A (16x1024 bf16, 32KB) staged in LDS once per block. On gfx1250 toolchains
// with the TDM builtin, the staging is a single Tensor-Data-Mover 1-D load
// (D# per ISA ch.8) issued by wave 0 + s_wait_tensorcnt; else a copy loop.
// grid = 250 blocks x 256 thr -> 2000 N-tiles.
// ---------------------------------------------------------------------------
__global__ void k_fc(const u16* __restrict__ A, const u16* __restrict__ Bw,
                     const float* __restrict__ bias, float* __restrict__ C)
{
  __shared__ __align__(16) u16 Ash[16 * Hd];
#if USE_TDM
  if (threadIdx.x < 32) {
    const unsigned long long ga = (unsigned long long)(uintptr_t)A;
    const unsigned lds_off = (unsigned)(uintptr_t)(void*)Ash;   // LDS aperture: low 32 bits
    const unsigned n = 16 * Hd;                                  // elements (2B each)
    // D# group0: count=1 | lds_addr | global_addr[56:0] | type=2
    v4u g0 = { 1u, lds_off,
               (unsigned)(ga & 0xFFFFFFFFu),
               (unsigned)((ga >> 32) & 0x01FFFFFFu) | 0x80000000u };
    // D# group1: data_size=2B; tensor_dim0=n; tensor_dim1=1; tile=(n,1);
    //            tensor_dim0_stride=n
    v8i g1 = { (int)0x00010000u,
               (int)((n & 0xFFFFu) << 16),
               (int)(((n >> 16) & 0xFFFFu) | (1u << 16)),
               (int)((n & 0xFFFFu) << 16),
               1, (int)n, 0, 0 };
    v4i g2 = { 0, 0, 0, 0 };
    v4i g3 = { 0, 0, 0, 0 };
#if __clang_major__ >= 23
    v8i g4 = { 0, 0, 0, 0, 0, 0, 0, 0 };
    __builtin_amdgcn_tensor_load_to_lds(g0, g1, g2, g3, g4, 0);
#else
    __builtin_amdgcn_tensor_load_to_lds(g0, g1, g2, g3, 0);
#endif
    __builtin_amdgcn_s_wait_tensorcnt(0);
  }
  __syncthreads();
#else
  {
    const u32x4* src = (const u32x4*)A;
    u32x4* dst = (u32x4*)Ash;
    for (int i = threadIdx.x; i < (16 * Hd) / 8; i += blockDim.x) dst[i] = src[i];
  }
  __syncthreads();
#endif
  const int lane = threadIdx.x & 31, wave = threadIdx.x >> 5;
  const int tile = blockIdx.x * 8 + wave;
  const int n0 = tile * 16;
  if (tile + 8 < Vd / 16)   // pull next block-row of weights toward L2/L0
    __builtin_prefetch(Bw + (size_t)(n0 + 128) * Hd, 0, 1);
  v8f acc = {};
  acc = wmma_tile_acc(Ash, Hd, Bw + (size_t)n0 * Hd, Hd, Hd, acc, lane);
  store_tile(acc, C, Vd, 0, n0, bias, lane);
}

// scores[b,s] = sum_h relu(x_proj[b,s,h] + hWh[b,h]) * w2[h] + b2
__global__ void k_attn_scores(const float* __restrict__ xproj,
                              const float* __restrict__ hWh,
                              const float* __restrict__ w2,
                              const float* __restrict__ b2,
                              float* __restrict__ scores)
{
  const int bs = blockIdx.x;               // 0 .. B*S-1
  const int b  = bs >> 7;
  const float* xp = xproj + (size_t)bs * Hd;
  const float* hw = hWh + (size_t)b * Hd;
  float p = 0.f;
  for (int hh = threadIdx.x; hh < Hd; hh += blockDim.x) {
    const float v = xp[hh] + hw[hh];
    p += (v > 0.f ? v : 0.f) * w2[hh];
  }
  __shared__ float red[256];
  red[threadIdx.x] = p; __syncthreads();
  for (int s = 128; s > 0; s >>= 1) {
    if (threadIdx.x < s) red[threadIdx.x] += red[threadIdx.x + s];
    __syncthreads();
  }
  if (threadIdx.x == 0) scores[bs] = red[0] + b2[0];
}

// log-softmax over S (faithful to reference!) then context -> bf16 (A1b of L0)
__global__ void k_softmax_ctx(const float* __restrict__ scores,
                              const float* __restrict__ x,
                              u16* __restrict__ ctx_bf)
{
  const int b = blockIdx.x;
  __shared__ float w[Sd];
  __shared__ float mx_s, ls_s;
  if (threadIdx.x < Sd) w[threadIdx.x] = scores[b * Sd + threadIdx.x];
  __syncthreads();
  if (threadIdx.x == 0) {
    float mx = -1e30f;
    for (int s = 0; s < Sd; ++s) mx = fmaxf(mx, w[s]);
    float sum = 0.f;
    for (int s = 0; s < Sd; ++s) sum += __expf(w[s] - mx);
    mx_s = mx; ls_s = logf(sum);
  }
  __syncthreads();
  if (threadIdx.x < Sd) w[threadIdx.x] = w[threadIdx.x] - mx_s - ls_s;
  __syncthreads();
  for (int e = threadIdx.x; e < Ed; e += blockDim.x) {
    float acc = 0.f;
    for (int s = 0; s < Sd; ++s)
      acc += w[s] * x[((size_t)(b * Sd + s)) * Ed + e];
    ctx_bf[b * Ed + e] = f2bf(acc);
  }
}

// final log-softmax over V=32000, writes out[b,t,:]
__global__ void k_logsoftmax_out(const float* __restrict__ logits,
                                 float* __restrict__ out, int t)
{
  const int b = blockIdx.x;
  const float* lg = logits + (size_t)b * Vd;
  __shared__ float red[256];
  float mx = -1e30f;
  for (int v = threadIdx.x; v < Vd; v += blockDim.x) mx = fmaxf(mx, lg[v]);
  red[threadIdx.x] = mx; __syncthreads();
  for (int s = 128; s > 0; s >>= 1) {
    if (threadIdx.x < s) red[threadIdx.x] = fmaxf(red[threadIdx.x], red[threadIdx.x + s]);
    __syncthreads();
  }
  mx = red[0]; __syncthreads();
  float sum = 0.f;
  for (int v = threadIdx.x; v < Vd; v += blockDim.x) sum += __expf(lg[v] - mx);
  red[threadIdx.x] = sum; __syncthreads();
  for (int s = 128; s > 0; s >>= 1) {
    if (threadIdx.x < s) red[threadIdx.x] += red[threadIdx.x + s];
    __syncthreads();
  }
  const float lse = mx + logf(red[0]);
  float* ob = out + ((size_t)b * Td + t) * Vd;
  for (int v = threadIdx.x; v < Vd; v += blockDim.x) ob[v] = lg[v] - lse;
}

// -------------------- one-time conversion / init kernels --------------------
__global__ void k_f2bf(const float* __restrict__ in, u16* __restrict__ out, long n)
{
  long i = (long)blockIdx.x * blockDim.x + threadIdx.x;
  if (i < n) out[i] = f2bf(in[i]);
}
// out[c*rows_in + r] = bf16(in[r*cols_in + c])  (transpose+convert)
__global__ void k_f2bf_T(const float* __restrict__ in, u16* __restrict__ out,
                         int rows_in, int cols_in)
{
  long i = (long)blockIdx.x * blockDim.x + threadIdx.x;
  long n = (long)rows_in * cols_in;
  if (i >= n) return;
  int c = (int)(i / rows_in), r = (int)(i % rows_in);
  out[(size_t)c * rows_in + r] = f2bf(in[(size_t)r * cols_in + c]);
}
// embT[t][b][e] = bf16(table[target[b][t]][e])  (time-major teacher inputs)
__global__ void k_embed(const int* __restrict__ tgt, const float* __restrict__ tbl,
                        u16* __restrict__ embT)
{
  long i = (long)blockIdx.x * blockDim.x + threadIdx.x;
  long n = (long)Td * Bd * Ed;
  if (i >= n) return;
  int t = (int)(i / (Bd * Ed));
  int b = (int)((i / Ed) % Bd);
  int e = (int)(i % Ed);
  int tok = tgt[b * Td + t];
  embT[i] = f2bf(tbl[(size_t)tok * Ed + e]);
}
__global__ void k_zero_f(float* p, long n)
{ long i = (long)blockIdx.x * blockDim.x + threadIdx.x; if (i < n) p[i] = 0.f; }
__global__ void k_zero_u(u16* p, long n)
{ long i = (long)blockIdx.x * blockDim.x + threadIdx.x; if (i < n) p[i] = 0; }

// ---------------------------------------------------------------------------
extern "C" void kernel_launch(void* const* d_in, const int* in_sizes, int n_in,
                              void* d_out, int out_size, void* d_ws, size_t ws_size,
                              hipStream_t stream) {
  (void)in_sizes; (void)n_in; (void)out_size; (void)ws_size;
  const float* x       = (const float*)d_in[0];
  const int*   tgt     = (const int*)d_in[1];
  const float* emb_tab = (const float*)d_in[2];
  const float* attn_Wx = (const float*)d_in[3];
  const float* attn_Wh = (const float*)d_in[4];
  const float* attn_b1 = (const float*)d_in[5];
  const float* attn_w2 = (const float*)d_in[6];
  const float* attn_b2 = (const float*)d_in[7];
  const float* fc_W    = (const float*)d_in[8];
  const float* fc_b    = (const float*)d_in[9];
  const float* Wih[3]  = {(const float*)d_in[10], (const float*)d_in[14], (const float*)d_in[18]};
  const float* Whh[3]  = {(const float*)d_in[11], (const float*)d_in[15], (const float*)d_in[19]};
  const float* bih[3]  = {(const float*)d_in[12], (const float*)d_in[16], (const float*)d_in[20]};
  const float* bhh[3]  = {(const float*)d_in[13], (const float*)d_in[17], (const float*)d_in[21]};
  float* out = (float*)d_out;

  // ---- workspace carve-out (256B aligned) ----
  char* base = (char*)d_ws;
  size_t off = 0;
  auto take = [&](size_t bytes) -> char* {
    char* p = base + off;
    off = (off + bytes + 255) & ~(size_t)255;
    return p;
  };
  u16* Xbf      = (u16*)take((size_t)Bd * Sd * Ed * 2);
  u16* WxT      = (u16*)take((size_t)Hd * Ed * 2);
  u16* WhT      = (u16*)take((size_t)Hd * Hd * 2);
  u16* Wihbf[3], *Whhbf[3];
  Wihbf[0] = (u16*)take((size_t)Gd * (2 * Ed) * 2);
  Whhbf[0] = (u16*)take((size_t)Gd * Hd * 2);
  Wihbf[1] = (u16*)take((size_t)Gd * Hd * 2);
  Whhbf[1] = (u16*)take((size_t)Gd * Hd * 2);
  Wihbf[2] = (u16*)take((size_t)Gd * Hd * 2);
  Whhbf[2] = (u16*)take((size_t)Gd * Hd * 2);
  u16*   fcWbf  = (u16*)take((size_t)Vd * Hd * 2);
  u16*   embT   = (u16*)take((size_t)Td * Bd * Ed * 2);
  float* xproj  = (float*)take((size_t)Bd * Sd * Hd * 4);
  float* hWh    = (float*)take((size_t)Bd * Hd * 4);
  float* scores = (float*)take((size_t)Bd * Sd * 4);
  u16*   ctx    = (u16*)take((size_t)Bd * Ed * 2);
  float* cst    = (float*)take((size_t)3 * Bd * Hd * 4);
  u16*   hping  = (u16*)take((size_t)3 * Bd * Hd * 2);   // h(t-1) / h(t) ping-pong
  u16*   hpong  = (u16*)take((size_t)3 * Bd * Hd * 2);
  float* logits = (float*)take((size_t)Bd * Vd * 4);

  auto g1 = [](long n) { return dim3((unsigned)((n + 255) / 256)); };
  const int BH = Bd * Hd;

  // ---- one-time: bf16 conversions, transposes, embeddings, state init ----
  { long n = (long)Bd * Sd * Ed;      k_f2bf<<<g1(n), 256, 0, stream>>>(x, Xbf, n); }
  { long n = (long)Ed * Hd;           k_f2bf_T<<<g1(n), 256, 0, stream>>>(attn_Wx, WxT, Ed, Hd); }
  { long n = (long)Hd * Hd;           k_f2bf_T<<<g1(n), 256, 0, stream>>>(attn_Wh, WhT, Hd, Hd); }
  { long n = (long)Gd * 2 * Ed;       k_f2bf<<<g1(n), 256, 0, stream>>>(Wih[0], Wihbf[0], n); }
  { long n = (long)Gd * Hd;           k_f2bf<<<g1(n), 256, 0, stream>>>(Whh[0], Whhbf[0], n); }
  { long n = (long)Gd * Hd;           k_f2bf<<<g1(n), 256, 0, stream>>>(Wih[1], Wihbf[1], n); }
  { long n = (long)Gd * Hd;           k_f2bf<<<g1(n), 256, 0, stream>>>(Whh[1], Whhbf[1], n); }
  { long n = (long)Gd * Hd;           k_f2bf<<<g1(n), 256, 0, stream>>>(Wih[2], Wihbf[2], n); }
  { long n = (long)Gd * Hd;           k_f2bf<<<g1(n), 256, 0, stream>>>(Whh[2], Whhbf[2], n); }
  { long n = (long)Vd * Hd;           k_f2bf<<<g1(n), 256, 0, stream>>>(fc_W, fcWbf, n); }
  { long n = (long)Td * Bd * Ed;      k_embed<<<g1(n), 256, 0, stream>>>(tgt, emb_tab, embT); }
  { long n = (long)3 * BH;            k_zero_f<<<g1(n), 256, 0, stream>>>(cst, n);
                                      k_zero_u<<<g1(n), 256, 0, stream>>>(hping, n);
                                      k_zero_u<<<g1(n), 256, 0, stream>>>(hpong, n); }

  // ---- one-time: x_proj = x @ attn_Wx + b1   (B*S=2048 x K=768 x N=1024) ----
  k_gemm_bias<<<dim3(Bd * Sd / 16, Hd / 128), 256, 0, stream>>>(
      Xbf, Ed, WxT, Ed, attn_b1, xproj, Hd, Ed);

  // ---- serial decode loop (8 launches/step) ----
  u16* hR = hping;   // holds h(t-1)
  u16* hW = hpong;   // receives h(t)
  for (int t = 0; t < Td; ++t) {
    // hWh = h2(t-1) @ attn_Wh        (16 x 1024 x 1024)
    k_gemm_bias<<<dim3(1, Hd / 128), 256, 0, stream>>>(
        hR + 2 * BH, Hd, WhT, Hd, nullptr, hWh, Hd, Hd);
    // attention scores + log-softmax weights + context (bf16)
    k_attn_scores<<<Bd * Sd, 256, 0, stream>>>(xproj, hWh, attn_w2, attn_b2, scores);
    k_softmax_ctx<<<Bd, 256, 0, stream>>>(scores, x, ctx);
    // LSTM layer 0: [emb_t | context] (16x1536) + h0(t-1) recurrence, fused act
    k_lstm_fused<<<Hd / 128, 256, 0, stream>>>(
        embT + (size_t)t * Bd * Ed, Ed, ctx, Ed, Wihbf[0], 2 * Ed,
        hR + 0 * BH, Whhbf[0], bih[0], bhh[0], cst + 0 * BH, hW + 0 * BH);
    // LSTM layer 1
    k_lstm_fused<<<Hd / 128, 256, 0, stream>>>(
        hW + 0 * BH, Hd, nullptr, 0, Wihbf[1], Hd,
        hR + 1 * BH, Whhbf[1], bih[1], bhh[1], cst + 1 * BH, hW + 1 * BH);
    // LSTM layer 2
    k_lstm_fused<<<Hd / 128, 256, 0, stream>>>(
        hW + 1 * BH, Hd, nullptr, 0, Wihbf[2], Hd,
        hR + 2 * BH, Whhbf[2], bih[2], bhh[2], cst + 2 * BH, hW + 2 * BH);
    // vocab projection + log-softmax
    k_fc<<<Vd / 128, 256, 0, stream>>>(hW + 2 * BH, fcWbf, fc_b, logits);
    k_logsoftmax_out<<<Bd, 256, 0, stream>>>(logits, out, t);
    // ping-pong h buffers
    u16* tmp = hR; hR = hW; hW = tmp;
  }
}
